// Hiremixer_45664092291371
// MI455X (gfx1250) — matrix operand order repaired and checked by reading, not compiled, not run on hardware
//
#include <hip/hip_runtime.h>
#include <hip/hip_bf16.h>
#include <math.h>

typedef _Float16 f16;
typedef __attribute__((ext_vector_type(16))) _Float16 v16h;
typedef __attribute__((ext_vector_type(8)))  float    v8f;

#define B_   2048
#define V_   17
#define C_   512
#define K_   3
#define L_   3
#define H_   16
#define D_   32
#define F_   ((size_t)B_ * V_ * C_)      // 17,825,792 tokens*chan
#define M_   (B_ * V_)                   // 34816 token rows
#define TOK_STRIDE (V_ * C_)             // 8704

// ---------------------------------------------------------------- helpers

__device__ __forceinline__ float gelu_f(float x) {
    return 0.5f * x * (1.0f + erff(x * 0.70710678118654752440f));
}

__device__ __forceinline__ v8f wmma_f16(v16h a, v16h b, v8f c) {
    return __builtin_amdgcn_wmma_f32_16x16x32_f16(false, a, false, b,
                                                  (short)0, c, false, false);
}

// A fragment 16x32 f16 (row-major source, row stride ld halves).
// lane = h*16+r : element e -> A[r][h*16 + e]
__device__ __forceinline__ v16h load_frag_a(const f16* __restrict__ base, int ld, int lane) {
    const int r = lane & 15, h = lane >> 4;
    const f16* p = base + (size_t)r * ld + h * 16;
    v16h out;
#pragma unroll
    for (int e = 0; e < 16; ++e) out[e] = p[e];
    return out;
}

// B fragment 32x16: B[k][n] = W[n][k], W row-major (row stride ld halves).
// lane = h*16+n : element e -> B[e + h*16][n]
__device__ __forceinline__ v16h load_frag_b(const f16* __restrict__ Wn0, int ld, int lane) {
    const int n = lane & 15, h = lane >> 4;
    const f16* p = Wn0 + (size_t)n * ld + h * 16;
    v16h out;
#pragma unroll
    for (int e = 0; e < 16; ++e) out[e] = p[e];
    return out;
}

// ---------------------------------------------------------------- elementwise

__global__ void __launch_bounds__(256) cvt_f16_kernel(const float* __restrict__ in,
                                                      f16* __restrict__ out, size_t n) {
    size_t i = (size_t)blockIdx.x * 256 + threadIdx.x;
    if (i < n) out[i] = (f16)in[i];
}

__global__ void __launch_bounds__(256) xin_kernel(const float* __restrict__ x,
                                                  float* __restrict__ out, size_t n) {
    size_t i = (size_t)blockIdx.x * 256 + threadIdx.x;
    if (i < n) { float v = x[i]; out[i] = v + gelu_f(v); }
}

__global__ void __launch_bounds__(256) add3_kernel(const float* __restrict__ a,
                                                   const float* __restrict__ b,
                                                   const float* __restrict__ c,
                                                   float* __restrict__ out, size_t n) {
    size_t i = (size_t)blockIdx.x * 256 + threadIdx.x;
    if (i < n) out[i] = a[i] + b[i] + c[i];
}

// ---------------------------------------------------------------- LayerNorms

// LN over C (512) per token row.  Optional fp32 out (may alias X) and/or f16 out.
__global__ void __launch_bounds__(128) ln_c_kernel(const float* __restrict__ X,
                                                   const float* __restrict__ g,
                                                   const float* __restrict__ b,
                                                   float* __restrict__ Yf,
                                                   f16* __restrict__ Yh) {
    __shared__ float red[128];
    const int tid = threadIdx.x;
    const size_t row = blockIdx.x;
    const float* x = X + row * C_;
    float s = 0.f, ss = 0.f;
#pragma unroll
    for (int c = tid; c < C_; c += 128) { float v = x[c]; s += v; ss += v * v; }
    red[tid] = s; __syncthreads();
    for (int o = 64; o > 0; o >>= 1) { if (tid < o) red[tid] += red[tid + o]; __syncthreads(); }
    float mean = red[0] * (1.0f / C_);
    __syncthreads();
    red[tid] = ss; __syncthreads();
    for (int o = 64; o > 0; o >>= 1) { if (tid < o) red[tid] += red[tid + o]; __syncthreads(); }
    float var = red[0] * (1.0f / C_) - mean * mean;
    float inv = rsqrtf(var + 1e-7f);
#pragma unroll
    for (int c = tid; c < C_; c += 128) {
        float v = (x[c] - mean) * inv * g[c] + b[c];
        if (Yf) Yf[row * C_ + c] = v;
        if (Yh) Yh[row * C_ + c] = (f16)v;
    }
}

// LN over J (17) per (b,c) strided column; f16 output only. g,b indexed by j.
__global__ void __launch_bounds__(256) ln_j_kernel(const float* __restrict__ X,
                                                   const float* __restrict__ g,
                                                   const float* __restrict__ b,
                                                   f16* __restrict__ Yh) {
    size_t t = (size_t)blockIdx.x * 256 + threadIdx.x;   // over B*C
    if (t >= (size_t)B_ * C_) return;
    const int c = (int)(t & (C_ - 1));
    const size_t bb = t >> 9;                            // /512
    const float* x = X + bb * TOK_STRIDE + c;
    float vals[V_];
    float s = 0.f, ss = 0.f;
#pragma unroll
    for (int j = 0; j < V_; ++j) { float v = x[(size_t)j * C_]; vals[j] = v; s += v; ss += v * v; }
    float mean = s * (1.0f / V_);
    float var  = ss * (1.0f / V_) - mean * mean;
    float inv  = rsqrtf(var + 1e-7f);
    f16* y = Yh + bb * TOK_STRIDE + c;
#pragma unroll
    for (int j = 0; j < V_; ++j)
        y[(size_t)j * C_] = (f16)((vals[j] - mean) * inv * g[j] + b[j]);
}

// ---------------------------------------------------------------- WMMA GEMM
// Y[M,N](f32) = X[M,512](f16) * W[N,512]^T(f16) + bias[N] (+ res[M,N]).
// grid = (M/256, N/64); block = 256 (8 waves); each wave does 32 rows x 64 cols.
__global__ void __launch_bounds__(256) gemm512_wmma(const f16* __restrict__ X,
                                                    const f16* __restrict__ W,
                                                    const float* __restrict__ bias,
                                                    const float* __restrict__ res,
                                                    float* __restrict__ Y,
                                                    int N) {
    const int lane = threadIdx.x & 31;
    const int wave = threadIdx.x >> 5;
    const size_t m0 = (size_t)blockIdx.x * 256 + (size_t)wave * 32;
    const int n0 = blockIdx.y * 64;

    v8f acc[2][4];
#pragma unroll
    for (int mm = 0; mm < 2; ++mm)
#pragma unroll
        for (int nn = 0; nn < 4; ++nn)
#pragma unroll
            for (int i = 0; i < 8; ++i) acc[mm][nn][i] = 0.f;

    const f16* x0 = X + m0 * C_;
    for (int k0 = 0; k0 < C_; k0 += 32) {
        v16h a0 = load_frag_a(x0 + k0, C_, lane);
        v16h a1 = load_frag_a(x0 + (size_t)16 * C_ + k0, C_, lane);
#pragma unroll
        for (int nn = 0; nn < 4; ++nn) {
            v16h bf = load_frag_b(W + (size_t)(n0 + nn * 16) * C_ + k0, C_, lane);
            acc[0][nn] = wmma_f16(a0, bf, acc[0][nn]);
            acc[1][nn] = wmma_f16(a1, bf, acc[1][nn]);
        }
    }

    const int nl = lane & 15, hh = lane >> 4;
#pragma unroll
    for (int mm = 0; mm < 2; ++mm)
#pragma unroll
        for (int nn = 0; nn < 4; ++nn) {
            const int col = n0 + nn * 16 + nl;
            const float bv = bias[col];
#pragma unroll
            for (int i = 0; i < 8; ++i) {
                const size_t row = m0 + mm * 16 + i + 8 * hh;
                const size_t idx = row * (size_t)N + col;
                float v = acc[mm][nn][i] + bv;
                if (res) v += res[idx];
                Y[idx] = v;
            }
        }
}

// ---------------------------------------------------------------- attention (N=17, D=32)
// QKV fp32 rows (b*17+v) x 1536 (s*512 + h*32 + d).  Out f16 token-major.
__global__ void __launch_bounds__(32) attn_kernel(const float* __restrict__ QKV,
                                                  f16* __restrict__ Out) {
    __shared__ float Ksh[V_][D_];
    __shared__ float Vsh[V_][D_];
    const int bh = blockIdx.x;
    const int b = bh >> 4, h = bh & 15;
    const float* base = QKV + (size_t)b * V_ * 1536 + h * D_;
    for (int idx = threadIdx.x; idx < V_ * D_; idx += 32) {
        int j = idx >> 5, d = idx & 31;
        Ksh[j][d] = base[(size_t)j * 1536 + 512 + d];
        Vsh[j][d] = base[(size_t)j * 1536 + 1024 + d];
    }
    __syncthreads();
    const int i = threadIdx.x;
    if (i < V_) {
        float q[D_];
#pragma unroll
        for (int d = 0; d < D_; ++d) q[d] = base[(size_t)i * 1536 + d];
        const float scale = 0.17677669529663687f;   // 32^-0.5
        float sc[V_], mx = -1e30f;
#pragma unroll
        for (int j = 0; j < V_; ++j) {
            float s = 0.f;
#pragma unroll
            for (int d = 0; d < D_; ++d) s += q[d] * Ksh[j][d];
            s *= scale; sc[j] = s; mx = fmaxf(mx, s);
        }
        float sum = 0.f;
#pragma unroll
        for (int j = 0; j < V_; ++j) { sc[j] = expf(sc[j] - mx); sum += sc[j]; }
        const float r = 1.0f / sum;
        f16* out = Out + (size_t)b * TOK_STRIDE + (size_t)i * C_ + h * D_;
#pragma unroll
        for (int d = 0; d < D_; ++d) {
            float o = 0.f;
#pragma unroll
            for (int j = 0; j < V_; ++j) o += sc[j] * Vsh[j][d];
            out[d] = (f16)(o * r);
        }
    }
}

// ---------------------------------------------------------------- GCN adjacency combine
// out[b,w,c] = res[b,w,c] + mlp[b,w,c] + sum_{k,v} z[b,v,k*512+c] * adj[k,v,w]
__global__ void __launch_bounds__(256) gcn_combine_kernel(const float* __restrict__ Z,
                                                          const float* __restrict__ adj,
                                                          const float* __restrict__ res,
                                                          const float* __restrict__ mlp,
                                                          float* __restrict__ out) {
    __shared__ float adj_s[K_ * V_ * V_];
    for (int i = threadIdx.x; i < K_ * V_ * V_; i += 256) adj_s[i] = adj[i];
    __syncthreads();
    size_t t = (size_t)blockIdx.x * 256 + threadIdx.x;   // over B*C
    const int c = (int)(t & (C_ - 1));
    const size_t bb = t >> 9;
    const float* z = Z + bb * ((size_t)V_ * 1536) + c;
    float acc[V_];
#pragma unroll
    for (int w = 0; w < V_; ++w) acc[w] = 0.f;
#pragma unroll
    for (int k = 0; k < K_; ++k)
#pragma unroll
        for (int v = 0; v < V_; ++v) {
            const float zv = z[(size_t)v * 1536 + k * C_];
            const float* a = adj_s + (k * V_ + v) * V_;
#pragma unroll
            for (int w = 0; w < V_; ++w) acc[w] += zv * a[w];
        }
    const size_t base = bb * TOK_STRIDE + c;
#pragma unroll
    for (int w = 0; w < V_; ++w) {
        const size_t idx = base + (size_t)w * C_;
        out[idx] = res[idx] + mlp[idx] + acc[w];
    }
}

// ---------------------------------------------------------------- fused per-batch MLP
// xn view A[c][j] (512x17, pad K->32); out[b,j,c] = gelu(gelu(A@W1^T+b1)@W2^T+b2)
// one block (8 waves) per batch; wave handles 64 c-rows; chained WMMA through LDS.

__device__ __forceinline__ v16h load_a_xpose(const f16* __restrict__ xn, int c0, int lane) {
    const int r = lane & 15, h = lane >> 4;
    const int c = c0 + r;
    v16h out;
#pragma unroll
    for (int e = 0; e < 16; ++e) {
        const int k = h * 16 + e;   // j index
        out[e] = (k < V_) ? xn[(size_t)k * C_ + c] : (f16)0;
    }
    return out;
}

__device__ __forceinline__ v16h load_b_w1(const f16* __restrict__ w1, int o0, int lane) {
    const int n = lane & 15, h = lane >> 4;
    const f16* p = w1 + (size_t)(o0 + n) * V_;
    v16h out;
#pragma unroll
    for (int e = 0; e < 16; ++e) { const int k = e + h * 16; out[e] = (k < V_) ? p[k] : (f16)0; }
    return out;
}

__device__ __forceinline__ v16h load_b_w2(const f16* __restrict__ w2, int j0, int oc, int lane) {
    const int n = lane & 15, h = lane >> 4;
    const int j = j0 + n;
    v16h out;
    if (j < V_) {
        const f16* p = w2 + (size_t)j * C_ + oc + h * 16;
#pragma unroll
        for (int e = 0; e < 16; ++e) out[e] = p[e];
    } else {
#pragma unroll
        for (int e = 0; e < 16; ++e) out[e] = (f16)0;
    }
    return out;
}

__global__ void __launch_bounds__(256) mlp_fused_kernel(const f16* __restrict__ XN,
                                                        const f16* __restrict__ W1h,
                                                        const float* __restrict__ b1,
                                                        const f16* __restrict__ W2h,
                                                        const float* __restrict__ b2,
                                                        float* __restrict__ Out) {
    __shared__ __align__(16) unsigned char scratch[8][4608];   // per-wave G (f16 64x32) / O (f32 64x17)
    const int b = blockIdx.x;
    const int lane = threadIdx.x & 31;
    const int wave = threadIdx.x >> 5;
    const int m0 = wave * 64;
    const int nl = lane & 15, hh = lane >> 4;
    const f16* xn = XN + (size_t)b * TOK_STRIDE;
    f16*   G = (f16*)(&scratch[wave][0]);
    float* O = (float*)(&scratch[wave][0]);

    v16h afrag[4];
#pragma unroll
    for (int mm = 0; mm < 4; ++mm) afrag[mm] = load_a_xpose(xn, m0 + mm * 16, lane);

    v8f acc2[4][2];
#pragma unroll
    for (int mm = 0; mm < 4; ++mm)
#pragma unroll
        for (int nn = 0; nn < 2; ++nn)
#pragma unroll
            for (int i = 0; i < 8; ++i) acc2[mm][nn][i] = 0.f;

    for (int oc = 0; oc < C_; oc += 32) {
        // GEMM1: G[64x32] = gelu(A[64x32] @ W1T[32x32] + b1)
        v8f g[4][2];
#pragma unroll
        for (int nn = 0; nn < 2; ++nn) {
            v16h bf = load_b_w1(W1h, oc + nn * 16, lane);
#pragma unroll
            for (int mm = 0; mm < 4; ++mm) {
                if (nn == 0) { for (int i = 0; i < 8; ++i) g[mm][0][i] = 0.f; }
                if (nn == 1) { for (int i = 0; i < 8; ++i) g[mm][1][i] = 0.f; }
                g[mm][nn] = wmma_f16(afrag[mm], bf, g[mm][nn]);
            }
        }
#pragma unroll
        for (int mm = 0; mm < 4; ++mm)
#pragma unroll
            for (int nn = 0; nn < 2; ++nn) {
                const int ol = nn * 16 + nl;
                const float bv = b1[oc + ol];
#pragma unroll
                for (int i = 0; i < 8; ++i) {
                    const int cl = mm * 16 + i + 8 * hh;
                    G[cl * 32 + ol] = (f16)gelu_f(g[mm][nn][i] + bv);
                }
            }
        __syncthreads();
        // GEMM2 accumulate: out2[64x32] += G[64x32] @ W2T[32x32]
        v16h bf0 = load_b_w2(W2h, 0, oc, lane);
        v16h bf1 = load_b_w2(W2h, 16, oc, lane);
#pragma unroll
        for (int mm = 0; mm < 4; ++mm) {
            v16h a2 = load_frag_a(G + mm * 16 * 32, 32, lane);
            acc2[mm][0] = wmma_f16(a2, bf0, acc2[mm][0]);
            acc2[mm][1] = wmma_f16(a2, bf1, acc2[mm][1]);
        }
        __syncthreads();
    }

    // bias + gelu, stage to LDS for coalesced writeout
#pragma unroll
    for (int mm = 0; mm < 4; ++mm)
#pragma unroll
        for (int nn = 0; nn < 2; ++nn) {
            const int j = nn * 16 + nl;
            if (j < V_) {
                const float bv = b2[j];
#pragma unroll
                for (int i = 0; i < 8; ++i) {
                    const int cl = mm * 16 + i + 8 * hh;
                    O[cl * V_ + j] = gelu_f(acc2[mm][nn][i] + bv);
                }
            }
        }
    __syncthreads();
    float* outb = Out + (size_t)b * TOK_STRIDE;
#pragma unroll
    for (int j = 0; j < V_; ++j)
        for (int cl = lane; cl < 64; cl += 32)
            outb[(size_t)j * C_ + m0 + cl] = O[cl * V_ + j];
}

// ---------------------------------------------------------------- launcher

extern "C" void kernel_launch(void* const* d_in, const int* in_sizes, int n_in,
                              void* d_out, int out_size, void* d_ws, size_t ws_size,
                              hipStream_t stream) {
    (void)in_sizes; (void)n_in; (void)out_size; (void)ws_size;

    const float* x_in     = (const float*)d_in[0];
    const float* adj      = (const float*)d_in[1];
    const float* ln1_g    = (const float*)d_in[2];
    const float* ln1_b    = (const float*)d_in[3];
    const float* ln2_g    = (const float*)d_in[4];
    const float* ln2_b    = (const float*)d_in[5];
    const float* gcn1_w   = (const float*)d_in[6];
    const float* gcn1_b   = (const float*)d_in[7];
    const float* gcn2_w   = (const float*)d_in[8];
    const float* gcn2_b   = (const float*)d_in[9];
    const float* m1w      = (const float*)d_in[10];
    const float* m1b      = (const float*)d_in[11];
    const float* m2w      = (const float*)d_in[12];
    const float* m2b      = (const float*)d_in[13];
    const float* aln_g    = (const float*)d_in[14];
    const float* aln_b    = (const float*)d_in[15];
    const float* qkv1_w   = (const float*)d_in[16];
    const float* qkv1_b   = (const float*)d_in[17];
    const float* proj1_w  = (const float*)d_in[18];
    const float* proj1_b  = (const float*)d_in[19];
    const float* qkv2_w   = (const float*)d_in[20];
    const float* qkv2_b   = (const float*)d_in[21];
    const float* proj2_w  = (const float*)d_in[22];
    const float* proj2_b  = (const float*)d_in[23];
    const float* tn_g     = (const float*)d_in[24];
    const float* tn_b     = (const float*)d_in[25];
    float* out = (float*)d_out;

    // workspace carve-out
    char* base = (char*)d_ws;
    size_t off = 0;
    auto take = [&](size_t bytes) -> void* {
        void* p = base + off;
        off += (bytes + 255) & ~(size_t)255;
        return p;
    };
    const size_t n_qkv = (size_t)L_ * 1536 * 512;   // 2,359,296
    const size_t n_prj = (size_t)L_ * 512 * 512;    //   786,432
    const size_t n_m   = (size_t)L_ * 512 * 17;     //    26,112
    f16* Wqkv1h  = (f16*)take(n_qkv * 2);
    f16* Wqkv2h  = (f16*)take(n_qkv * 2);
    f16* Wproj1h = (f16*)take(n_prj * 2);
    f16* Wproj2h = (f16*)take(n_prj * 2);
    f16* Wgcn1h  = (f16*)take(n_qkv * 2);
    f16* Wgcn2h  = (f16*)take(n_qkv * 2);
    f16* Wm1h    = (f16*)take(n_m * 2);
    f16* Wm2h    = (f16*)take(n_m * 2);
    float* A   = (float*)take(F_ * 4);                    // xin / x7
    float* T1  = (float*)take(F_ * 4);                    // mlp out
    float* T2  = (float*)take(F_ * 4);                    // x_mid / x2
    float* T3  = (float*)take(F_ * 4);                    // x5a / x5
    f16*   Hh  = (f16*)take(F_ * 2);                      // f16 staging
    float* Big = (float*)take((size_t)M_ * 1536 * 4);     // qkv / z

    auto cvt = [&](const float* src, f16* dst, size_t n) {
        cvt_f16_kernel<<<(unsigned)((n + 255) / 256), 256, 0, stream>>>(src, dst, n);
    };
    cvt(qkv1_w, Wqkv1h, n_qkv);   cvt(qkv2_w, Wqkv2h, n_qkv);
    cvt(proj1_w, Wproj1h, n_prj); cvt(proj2_w, Wproj2h, n_prj);
    cvt(gcn1_w, Wgcn1h, n_qkv);   cvt(gcn2_w, Wgcn2h, n_qkv);
    cvt(m1w, Wm1h, n_m);          cvt(m2w, Wm2h, n_m);

    const unsigned EW_GRID  = (unsigned)((F_ + 255) / 256);   // 69632
    const dim3 GEMM_QKV(136, 24), GEMM_PRJ(136, 8);

    for (int i = 0; i < L_; ++i) {
        const float* xcur = (i == 0) ? x_in : out;
        const size_t iq = (size_t)i * 1536 * 512, ip = (size_t)i * 512 * 512, im = (size_t)i * 8704;

        // xin = x + gelu(x)
        xin_kernel<<<EW_GRID, 256, 0, stream>>>(xcur, A, F_);

        // ---- gkpc ----
        ln_c_kernel<<<M_, 128, 0, stream>>>(A, aln_g + i * 512, aln_b + i * 512, nullptr, Hh);
        gemm512_wmma<<<GEMM_QKV, 256, 0, stream>>>(Hh, Wqkv1h + iq, qkv1_b + i * 1536, nullptr, Big, 1536);
        attn_kernel<<<B_ * H_, 32, 0, stream>>>(Big, Hh);
        gemm512_wmma<<<GEMM_PRJ, 256, 0, stream>>>(Hh, Wproj1h + ip, proj1_b + i * 512, A, T3, 512);   // x5a
        ln_c_kernel<<<M_, 128, 0, stream>>>(T3, aln_g + i * 512, aln_b + i * 512, nullptr, Hh);
        gemm512_wmma<<<GEMM_QKV, 256, 0, stream>>>(Hh, Wqkv2h + iq, qkv2_b + i * 1536, nullptr, Big, 1536);
        attn_kernel<<<B_ * H_, 32, 0, stream>>>(Big, Hh);
        gemm512_wmma<<<GEMM_PRJ, 256, 0, stream>>>(Hh, Wproj2h + ip, proj2_b + i * 512, nullptr, T3, 512); // x5

        // ---- lrsc block 1 ----
        ln_j_kernel<<<4096, 256, 0, stream>>>(A, ln1_g + i * V_, ln1_b + i * V_, Hh);
        mlp_fused_kernel<<<B_, 256, 0, stream>>>(Hh, Wm1h + im, m1b + i * 512, Wm2h + im, m2b + i * V_, T1);
        gemm512_wmma<<<GEMM_QKV, 256, 0, stream>>>(Hh, Wgcn1h + iq, gcn1_b + i * 1536, nullptr, Big, 1536);
        gcn_combine_kernel<<<4096, 256, 0, stream>>>(Big, adj, A, T1, T2);   // x_mid
        // LN over C (in-place fp32 + f16)
        ln_c_kernel<<<M_, 128, 0, stream>>>(T2, ln2_g + i * 512, ln2_b + i * 512, T2, Hh);
        // ---- lrsc block 2 ----
        mlp_fused_kernel<<<B_, 256, 0, stream>>>(Hh, Wm1h + im, m1b + i * 512, Wm2h + im, m2b + i * V_, T1);
        gemm512_wmma<<<GEMM_QKV, 256, 0, stream>>>(Hh, Wgcn2h + iq, gcn2_b + i * 1536, nullptr, Big, 1536);
        gcn_combine_kernel<<<4096, 256, 0, stream>>>(Big, adj, T2, T1, A);   // x7

        // x = x + x5 + x7
        add3_kernel<<<EW_GRID, 256, 0, stream>>>(xcur, T3, A, out, F_);
    }

    // final LayerNorm over C, in place on d_out
    ln_c_kernel<<<M_, 128, 0, stream>>>(out, tn_g, tn_b, out, nullptr);
}
